// SerriformNet_10548439679756
// MI455X (gfx1250) — compile-verified, matrix operand order checked by reading, not run on hardware
//
#include <hip/hip_runtime.h>
#include <hip/hip_bf16.h>
#include <cmath>

// ---------------------------------------------------------------------------
// SerriformNet forward, MI455X (gfx1250, wave32, WMMA).
// All dense projections run through one bf16-WMMA GEMM (TN: A[M,K] * B[N,K]^T),
// fp32 accumulate, fused bias + activation epilogue.
// GEMM: 128x64 block tile, K-step 64, float4 register-pipelined staging,
// ds_load_b128 fragment feeds, optional non-temporal C stores (logits).
// ---------------------------------------------------------------------------

static constexpr int Lc  = 4;
static constexpr int Dc  = 512;
static constexpr int DMc = 128;
static constexpr int Ec  = 2;
static constexpr int DHc = 128;
static constexpr int Kc  = 5;
static constexpr int Vc  = 50257;
static constexpr int Bc  = 2;
static constexpr int Sc  = 1024;
static constexpr int BSc = Bc * Sc;
static constexpr float EPSc = 1e-6f;

typedef __attribute__((ext_vector_type(16))) __bf16 v16bf;
typedef __attribute__((ext_vector_type(8)))  float  v8f;

// ------------------------------ helpers ------------------------------------

__device__ __forceinline__ float apply_act(float x, int act) {
  if (act == 1) return x / (1.0f + __expf(-x));                       // SiLU
  if (act == 2) return 0.5f * x * (1.0f + erff(x * 0.70710678118f));  // exact GELU
  return x;
}

// fp32 -> bf16 round-to-nearest-even, pure 32-bit integer VALU (no 16-bit
// subregister churn; co-executes under WMMA).
__device__ __forceinline__ unsigned bf16_rne(float f) {
  const unsigned u = __builtin_bit_cast(unsigned, f);
  return (u + 0x7FFFu + ((u >> 16) & 1u)) >> 16;
}
__device__ __forceinline__ uint2 pack4_bf16(float4 f) {
  uint2 r;
  r.x = bf16_rne(f.x) | (bf16_rne(f.y) << 16);
  r.y = bf16_rne(f.z) | (bf16_rne(f.w) << 16);
  return r;
}

// block reduction (blockDim.x multiple of 32, <= 256)
__device__ __forceinline__ float block_reduce_sum(float v, float* red) {
  const int lane = threadIdx.x & 31;
  const int wv   = threadIdx.x >> 5;
  const int nw   = blockDim.x >> 5;
  #pragma unroll
  for (int off = 16; off; off >>= 1) v += __shfl_xor(v, off);
  if (lane == 0) red[wv] = v;
  __syncthreads();
  if (wv == 0) {
    float t = (lane < nw) ? red[lane] : 0.0f;
    #pragma unroll
    for (int off = 8; off; off >>= 1) t += __shfl_xor(t, off);
    if (lane == 0) red[0] = t;
  }
  __syncthreads();
  return red[0];
}

// ------------------------------ WMMA GEMM ----------------------------------
static constexpr int TM = 128;
static constexpr int TN = 64;
static constexpr int TK = 64;            // two 16x16x32 k-slabs per stage
static constexpr int AQ = (TM * TK / 4) / 256;  // float4 gathers per thread (A) = 8
static constexpr int BQ = (TN * TK / 4) / 256;  // float4 gathers per thread (B) = 4
static constexpr int LDA_S = TK + 8;     // LDS stride (bf16 elems), 8B-aligned + skew
static constexpr int LDB_S = TK + 8;

__device__ __forceinline__ void gather_tiles(
    const float* __restrict__ A, int lda,
    const float* __restrict__ Bm, int ldb,
    int blockM, int blockN, int N, int k0, int tid,
    float4 (&aR)[AQ], float4 (&bR)[BQ])
{
  #pragma unroll
  for (int it = 0; it < AQ; ++it) {
    const int idx = it * 256 + tid;          // quad index in 128x(TK/4)
    const int row = idx >> 4;                // 16 quads per row
    const int col = (idx & 15) * 4;
    aR[it] = *(const float4*)&A[(size_t)(blockM + row) * lda + (k0 + col)];
  }
  #pragma unroll
  for (int it = 0; it < BQ; ++it) {
    const int idx = it * 256 + tid;
    const int row = idx >> 4;
    const int col = (idx & 15) * 4;
    const int n = blockN + row;
    bR[it] = (n < N) ? *(const float4*)&Bm[(size_t)n * ldb + (k0 + col)]
                     : make_float4(0.f, 0.f, 0.f, 0.f);
  }
}

__global__ __launch_bounds__(256)
void gemm_bf16_wmma(const float* __restrict__ A, int lda,
                    const float* __restrict__ Bm, int ldb,
                    const float* __restrict__ bias,
                    float* __restrict__ C, int ldc,
                    int M, int N, int Kd, int act, int nt)
{
  __shared__ __bf16 As[TM * LDA_S];
  __shared__ __bf16 Bs[TN * LDB_S];

  const int tid  = threadIdx.x;
  const int lane = tid & 31;
  const int wv   = tid >> 5;     // 0..7
  const int wm   = wv >> 1;      // 0..3 (M quadrant)
  const int wn   = wv & 1;       // 0..1 (N half)

  const int blockM = blockIdx.y * TM;   // M is always a multiple of 128 here
  const int blockN = blockIdx.x * TN;

  const int lr    = lane & 15;
  const int khalf = (lane >> 4) * 8;    // ISA 16-bit A/B layout: lane halves

  v8f acc[2][2] = {};

  float4 aR[AQ], bR[BQ];
  gather_tiles(A, lda, Bm, ldb, blockM, blockN, N, 0, tid, aR, bR);

  for (int k0 = 0; k0 < Kd; k0 += TK) {
    __syncthreads();   // previous stage's fragment reads done (WAR)
    // ---- store staged registers to LDS as packed bf16 (ds_store_b64) ----
    #pragma unroll
    for (int it = 0; it < AQ; ++it) {
      const int idx = it * 256 + tid;
      const int row = idx >> 4;
      const int col = (idx & 15) * 4;
      *(uint2*)&As[row * LDA_S + col] = pack4_bf16(aR[it]);
    }
    #pragma unroll
    for (int it = 0; it < BQ; ++it) {
      const int idx = it * 256 + tid;
      const int row = idx >> 4;
      const int col = (idx & 15) * 4;
      *(uint2*)&Bs[row * LDB_S + col] = pack4_bf16(bR[it]);
    }
    __syncthreads();

    // ---- software pipeline: gather next K-slab while computing this one ----
    if (k0 + TK < Kd)
      gather_tiles(A, lda, Bm, ldb, blockM, blockN, N, k0 + TK, tid, aR, bR);

    // ---- two 32-wide k-slabs: build fragments + 8x v_wmma per stage ----
    #pragma unroll
    for (int ks = 0; ks < 2; ++ks) {
      const int kc = ks * 32;
      v16bf afrag[2], bfrag[2];
      #pragma unroll
      for (int mi = 0; mi < 2; ++mi) {
        const int m = wm * 32 + mi * 16 + lr;
        #pragma unroll
        for (int vv = 0; vv < 8; ++vv) {
          const int kb = kc + ((vv < 4) ? (2 * vv) : (16 + 2 * (vv - 4))) + khalf;
          afrag[mi][2 * vv]     = As[m * LDA_S + kb];
          afrag[mi][2 * vv + 1] = As[m * LDA_S + kb + 1];
        }
      }
      #pragma unroll
      for (int ni = 0; ni < 2; ++ni) {
        const int n = wn * 32 + ni * 16 + lr;
        #pragma unroll
        for (int vv = 0; vv < 8; ++vv) {
          const int kb = kc + ((vv < 4) ? (2 * vv) : (16 + 2 * (vv - 4))) + khalf;
          bfrag[ni][2 * vv]     = Bs[n * LDB_S + kb];
          bfrag[ni][2 * vv + 1] = Bs[n * LDB_S + kb + 1];
        }
      }
      #pragma unroll
      for (int mi = 0; mi < 2; ++mi)
        #pragma unroll
        for (int ni = 0; ni < 2; ++ni)
          acc[mi][ni] = __builtin_amdgcn_wmma_f32_16x16x32_bf16(
              false, afrag[mi], false, bfrag[ni],
              (short)0, acc[mi][ni], false, false);
    }
  }

  // ---- epilogue: bias + activation; NT stores for streaming output ----
  const int mrow0 = (lane >= 16) ? 8 : 0;
  #pragma unroll
  for (int mi = 0; mi < 2; ++mi) {
    #pragma unroll
    for (int ni = 0; ni < 2; ++ni) {
      const int n = blockN + wn * 32 + ni * 16 + lr;
      if (n >= N) continue;
      const float bv = bias ? bias[n] : 0.0f;
      #pragma unroll
      for (int i = 0; i < 8; ++i) {
        const int m = blockM + wm * 32 + mi * 16 + mrow0 + i;
        const float val = apply_act(acc[mi][ni][i] + bv, act);
        if (nt)
          __builtin_nontemporal_store(val, &C[(size_t)m * ldc + n]);
        else
          C[(size_t)m * ldc + n] = val;
      }
    }
  }
}

// ------------------------- elementwise / small kernels ----------------------

__global__ void embed_rope_kernel(const int* __restrict__ tokens,
                                  const float* __restrict__ embed,
                                  float* __restrict__ x)
{
  const int t   = blockIdx.x;           // 0..BS-1
  const int s   = t % Sc;
  const int tok = tokens[t];
  const float* row = embed + (size_t)tok * Dc;
  for (int d = threadIdx.x; d < Dc; d += blockDim.x) {
    const int idx  = (d < Dc / 2) ? d : d - Dc / 2;
    const float inv = __powf(10000.0f, -(float)idx / (float)(Dc / 2));
    const float ang = (float)s * inv;
    const float rot = (d < Dc / 2) ? -row[d + Dc / 2] : row[d - Dc / 2];
    x[(size_t)t * Dc + d] = row[d] * __cosf(ang) + rot * __sinf(ang);
  }
}

__global__ __launch_bounds__(256)
void rmsnorm_kernel(const float* __restrict__ in, const float* __restrict__ w,
                    float* __restrict__ out)
{
  __shared__ float red[8];
  const int row = blockIdx.x;
  float ss = 0.0f;
  for (int d = threadIdx.x; d < Dc; d += 256) {
    const float v = in[(size_t)row * Dc + d];
    ss += v * v;
  }
  const float tot = block_reduce_sum(ss, red);
  const float rs  = rsqrtf(tot / (float)Dc + EPSc);
  for (int d = threadIdx.x; d < Dc; d += 256)
    out[(size_t)row * Dc + d] = w[d] * in[(size_t)row * Dc + d] * rs;
}

__global__ void dwconv_kernel(const float* __restrict__ xn,
                              const float* __restrict__ w,   // [D,K]
                              const float* __restrict__ bias,
                              float* __restrict__ y, int dil)
{
  const int t = blockIdx.x;
  const int b = t / Sc, s = t % Sc;
  for (int d = threadIdx.x; d < Dc; d += blockDim.x) {
    float acc = bias[d];
    #pragma unroll
    for (int k = 0; k < Kc; ++k) {
      const int sp = s - (Kc - 1 - k) * dil;
      if (sp >= 0)
        acc += w[d * Kc + k] * xn[((size_t)b * Sc + sp) * Dc + d];
    }
    y[(size_t)t * Dc + d] = acc;
  }
}

__global__ __launch_bounds__(128)
void router_kernel(const float* __restrict__ h, const float* __restrict__ rt_w,
                   const float* __restrict__ rt_b, float* __restrict__ r)
{
  __shared__ float red[8];
  const int t = blockIdx.x;
  float a0 = 0.0f, a1 = 0.0f;
  for (int d = threadIdx.x; d < Dc; d += 128) {
    const float hv = h[(size_t)t * Dc + d];
    a0 += hv * rt_w[d];
    a1 += hv * rt_w[Dc + d];
  }
  const float l0 = block_reduce_sum(a0, red) + rt_b[0];
  __syncthreads();
  const float l1 = block_reduce_sum(a1, red) + rt_b[1];
  if (threadIdx.x == 0) {
    const float mx = fmaxf(l0, l1);
    const float e0 = __expf(l0 - mx), e1 = __expf(l1 - mx);
    const float inv = 1.0f / (e0 + e1);
    r[(size_t)t * Ec + 0] = e0 * inv;
    r[(size_t)t * Ec + 1] = e1 * inv;
  }
}

__global__ __launch_bounds__(256)
void combine_rmsnorm_kernel(const float* __restrict__ h,
                            const float* __restrict__ r,
                            const float* __restrict__ eo,   // [BS,E,D]
                            const float* __restrict__ w,
                            float* __restrict__ out)
{
  __shared__ float vbuf[Dc];
  __shared__ float red[8];
  const int t = blockIdx.x;
  const float r0 = r[(size_t)t * Ec + 0], r1 = r[(size_t)t * Ec + 1];
  float ss = 0.0f;
  for (int d = threadIdx.x; d < Dc; d += 256) {
    const float v = h[(size_t)t * Dc + d]
                  + r0 * eo[(size_t)t * Ec * Dc + d]
                  + r1 * eo[(size_t)t * Ec * Dc + Dc + d];
    vbuf[d] = v;
    ss += v * v;
  }
  const float tot = block_reduce_sum(ss, red);
  const float rs  = rsqrtf(tot / (float)Dc + EPSc);
  __syncthreads();
  for (int d = threadIdx.x; d < Dc; d += 256)
    out[(size_t)t * Dc + d] = w[d] * vbuf[d] * rs;
}

__global__ void add2_kernel(const float* a, const float* b, float* o, int n) {
  const int i = blockIdx.x * 256 + threadIdx.x;
  if (i < n) o[i] = a[i] + b[i];
}
__global__ void add3_kernel(const float* a, const float* b, const float* c,
                            float* o, int n) {
  const int i = blockIdx.x * 256 + threadIdx.x;
  if (i < n) o[i] = a[i] + b[i] + c[i];
}

// ------------------------- sequential recurrence scan -----------------------
__global__ __launch_bounds__(512)
void scan_kernel(const float* __restrict__ xn,
                 const float* __restrict__ fx, const float* __restrict__ ux,
                 const float* __restrict__ mu, const float* __restrict__ ox,
                 const float* __restrict__ wf,   // [DM, D+DM]
                 const float* __restrict__ wu,   // [DM, D+DM]
                 const float* __restrict__ wo,   // [D,  D+DM]
                 float* __restrict__ xr)
{
  const int b   = blockIdx.x;
  const int tid = threadIdx.x;
  __shared__ float m[DMc];
  if (tid < DMc) m[tid] = 0.0f;
  __syncthreads();

  for (int t = 0; t < Sc; ++t) {
    const size_t baseM = ((size_t)b * Sc + t) * DMc;
    const size_t baseD = ((size_t)b * Sc + t) * Dc;

    float newm = 0.0f;
    if (tid < DMc) {
      float df = fx[baseM + tid];
      float du = ux[baseM + tid];
      const float* wfr = wf + (size_t)tid * (Dc + DMc) + Dc;
      const float* wur = wu + (size_t)tid * (Dc + DMc) + Dc;
      #pragma unroll 4
      for (int j = 0; j < DMc; ++j) {
        const float mj = m[j];
        df += mj * wfr[j];
        du += mj * wur[j];
      }
      const float f = 1.0f / (1.0f + __expf(-df));
      const float u = 1.0f / (1.0f + __expf(-du));
      newm = f * m[tid] + u * mu[baseM + tid];
    }
    __syncthreads();
    if (tid < DMc) m[tid] = newm;
    __syncthreads();

    {
      const int o = tid;
      const float* wor = wo + (size_t)o * (Dc + DMc) + Dc;
      float acc = xn[baseD + o] + ox[baseD + o];
      #pragma unroll 4
      for (int j = 0; j < DMc; ++j) acc += m[j] * wor[j];
      xr[baseD + o] = acc;
    }
    __syncthreads();
  }
}

// ------------------------------- host glue ----------------------------------

static void launch_gemm(const float* A, int lda, const float* Bm, int ldb,
                        const float* bias, float* C, int ldc,
                        int M, int N, int Kd, int act, int nt, hipStream_t s)
{
  dim3 grid((N + TN - 1) / TN, M / TM);
  gemm_bf16_wmma<<<grid, 256, 0, s>>>(A, lda, Bm, ldb, bias, C, ldc,
                                      M, N, Kd, act, nt);
}

extern "C" void kernel_launch(void* const* d_in, const int* in_sizes, int n_in,
                              void* d_out, int out_size, void* d_ws, size_t ws_size,
                              hipStream_t stream)
{
  (void)in_sizes; (void)n_in; (void)out_size; (void)ws_size;

  const int*   tokens = (const int*)  d_in[0];
  const float* embed  = (const float*)d_in[1];
  const float* norm_w = (const float*)d_in[2];
  const float* dw_w   = (const float*)d_in[3];
  const float* dw_b   = (const float*)d_in[4];
  const float* pw_w   = (const float*)d_in[5];
  const float* pw_b   = (const float*)d_in[6];
  const float* rin_w  = (const float*)d_in[7];
  const float* rin_b  = (const float*)d_in[8];
  const float* rf_w   = (const float*)d_in[9];
  const float* rf_b   = (const float*)d_in[10];
  const float* ru_w   = (const float*)d_in[11];
  const float* ru_b   = (const float*)d_in[12];
  const float* ro_w   = (const float*)d_in[13];
  const float* ro_b   = (const float*)d_in[14];
  const float* rt_w   = (const float*)d_in[15];
  const float* rt_b   = (const float*)d_in[16];
  const float* e1_w   = (const float*)d_in[17];
  const float* e1_b   = (const float*)d_in[18];
  const float* e2_w   = (const float*)d_in[19];
  const float* e2_b   = (const float*)d_in[20];
  const float* fn_w   = (const float*)d_in[21];
  const float* fd_w   = (const float*)d_in[22];
  const float* fd_b   = (const float*)d_in[23];
  const float* fu_w   = (const float*)d_in[24];
  const float* fu_b   = (const float*)d_in[25];
  const float* fnorm  = (const float*)d_in[26];
  float* out = (float*)d_out;

  // workspace carve-up (floats)
  float* ws = (float*)d_ws;
  size_t off = 0;
  auto alloc = [&](size_t n) { float* p = ws + off; off += n; return p; };
  float* xA   = alloc((size_t)BSc * Dc);
  float* xB   = alloc((size_t)BSc * Dc);
  float* xn   = alloc((size_t)BSc * Dc);
  float* conv = alloc((size_t)BSc * Dc);
  float* xc   = alloc((size_t)BSc * Dc);
  float* fx   = alloc((size_t)BSc * DMc);
  float* ux   = alloc((size_t)BSc * DMc);
  float* muv  = alloc((size_t)BSc * DMc);
  float* oxv  = alloc((size_t)BSc * Dc);
  float* xr   = alloc((size_t)BSc * Dc);
  float* h    = alloc((size_t)BSc * Dc);
  float* rbuf = alloc((size_t)BSc * Ec);
  float* hh   = alloc((size_t)BSc * Ec * Dc);
  float* eo   = alloc((size_t)BSc * Ec * Dc);
  float* hn   = alloc((size_t)BSc * Dc);
  float* g    = alloc((size_t)BSc * DHc);
  float* u    = alloc((size_t)BSc * Dc);

  const int dils[Lc] = {1, 1, 2, 4};  // max(1, int(exp(i*log(8)/4)))

  embed_rope_kernel<<<BSc, 256, 0, stream>>>(tokens, embed, xA);

  float* xcur = xA;
  float* xnext = xB;
  const int nElem = BSc * Dc;
  const int eBlocks = (nElem + 255) / 256;

  for (int i = 0; i < Lc; ++i) {
    rmsnorm_kernel<<<BSc, 256, 0, stream>>>(xcur, norm_w + (size_t)i * Dc, xn);
    dwconv_kernel<<<BSc, 256, 0, stream>>>(xn, dw_w + (size_t)i * Dc * Kc,
                                           dw_b + (size_t)i * Dc, conv, dils[i]);
    // pointwise: xc = conv @ pw_w^T + pw_b
    launch_gemm(conv, Dc, pw_w + (size_t)i * Dc * Dc, Dc,
                pw_b + (size_t)i * Dc, xc, Dc, BSc, Dc, Dc, 0, 0, stream);
    // recurrence input projections
    launch_gemm(xn, Dc, rin_w + (size_t)i * DMc * Dc, Dc,
                rin_b + (size_t)i * DMc, muv, DMc, BSc, DMc, Dc, 1, 0, stream);
    launch_gemm(xn, Dc, rf_w + (size_t)i * DMc * (Dc + DMc), Dc + DMc,
                rf_b + (size_t)i * DMc, fx, DMc, BSc, DMc, Dc, 0, 0, stream);
    launch_gemm(xn, Dc, ru_w + (size_t)i * DMc * (Dc + DMc), Dc + DMc,
                ru_b + (size_t)i * DMc, ux, DMc, BSc, DMc, Dc, 0, 0, stream);
    launch_gemm(xn, Dc, ro_w + (size_t)i * Dc * (Dc + DMc), Dc + DMc,
                ro_b + (size_t)i * Dc, oxv, Dc, BSc, Dc, Dc, 0, 0, stream);
    scan_kernel<<<Bc, 512, 0, stream>>>(xn, fx, ux, muv, oxv,
        rf_w + (size_t)i * DMc * (Dc + DMc),
        ru_w + (size_t)i * DMc * (Dc + DMc),
        ro_w + (size_t)i * Dc * (Dc + DMc), xr);
    add2_kernel<<<eBlocks, 256, 0, stream>>>(xc, xr, h, nElem);
    router_kernel<<<BSc, 128, 0, stream>>>(h, rt_w + (size_t)i * Ec * Dc,
                                           rt_b + (size_t)i * Ec, rbuf);
    for (int e = 0; e < Ec; ++e) {
      const size_t we = ((size_t)i * Ec + e);
      launch_gemm(h, Dc, e1_w + we * Dc * Dc, Dc, e1_b + we * Dc,
                  hh + (size_t)e * Dc, Ec * Dc, BSc, Dc, Dc, 1, 0, stream);
      launch_gemm(hh + (size_t)e * Dc, Ec * Dc, e2_w + we * Dc * Dc, Dc,
                  e2_b + we * Dc, eo + (size_t)e * Dc, Ec * Dc,
                  BSc, Dc, Dc, 0, 0, stream);
    }
    combine_rmsnorm_kernel<<<BSc, 256, 0, stream>>>(h, rbuf, eo,
                                                    fn_w + (size_t)i * Dc, hn);
    launch_gemm(hn, Dc, fd_w + (size_t)i * DHc * Dc, Dc,
                fd_b + (size_t)i * DHc, g, DHc, BSc, DHc, Dc, 2, 0, stream);
    launch_gemm(g, DHc, fu_w + (size_t)i * Dc * DHc, DHc,
                fu_b + (size_t)i * Dc, u, Dc, BSc, Dc, DHc, 0, 0, stream);
    add3_kernel<<<eBlocks, 256, 0, stream>>>(hn, u, xcur, xnext, nElem);
    float* tmp = xcur; xcur = xnext; xnext = tmp;
  }

  rmsnorm_kernel<<<BSc, 256, 0, stream>>>(xcur, fnorm, xn);
  // logits = xn @ embed^T -> [BS, V]. Store-BW bound (~412 MB out); use
  // non-temporal C stores so the L2-resident embed table (~103 MB) survives.
  launch_gemm(xn, Dc, embed, Dc, nullptr, out, Vc, BSc, Vc, Dc, 0, 1, stream);
}
